// AdaptiveGRU_22849226014720
// MI455X (gfx1250) — matrix operand order repaired and checked by reading, not compiled, not run on hardware
//
#include <hip/hip_runtime.h>
#include <hip/hip_bf16.h>

#define H  1024
#define B  64
#define T  256
#define N3 3072

typedef __bf16 bf16_t;
typedef __bf16 v16bf __attribute__((ext_vector_type(16)));
typedef __bf16 v8bf  __attribute__((ext_vector_type(8)));
typedef float v8f    __attribute__((ext_vector_type(8)));

// ---------------- WMMA helpers (16x16x32 bf16, fp32 accumulate) ----------------

// A tile 16x32 (MxK): lane l<16 holds row l, K={0..7,16..23}; lane>=16 row l-16, K={8..15,24..31}
__device__ __forceinline__ v16bf load_a_tile(const bf16_t* __restrict__ p0, int ld, int lane) {
  int m  = lane & 15;
  int hi = lane >> 4;
  const bf16_t* p = p0 + m * ld + (hi ? 8 : 0);
  v8bf lo = *(const v8bf*)(p);
  v8bf hb = *(const v8bf*)(p + 16);
  return __builtin_shufflevector(lo, hb, 0,1,2,3,4,5,6,7,8,9,10,11,12,13,14,15);
}

// B tile 32x16 (KxN), weights stored row-major [N,K]: lane l<16 holds col l K={0..7,16..23}; lane>=16 col l-16 K={8..15,24..31}
__device__ __forceinline__ v16bf load_b_tile(const bf16_t* __restrict__ p0, int ld, int lane) {
  int n  = lane & 15;
  int hi = lane >> 4;
  const bf16_t* p = p0 + n * ld + (hi ? 8 : 0);
  v8bf lo = *(const v8bf*)(p);
  v8bf hb = *(const v8bf*)(p + 16);
  return __builtin_shufflevector(lo, hb, 0,1,2,3,4,5,6,7,8,9,10,11,12,13,14,15);
}

__device__ __forceinline__ v8f wmma_bf16(v16bf a, v16bf b, v8f c) {
  return __builtin_amdgcn_wmma_f32_16x16x32_bf16(false, a, false, b, (short)0, c, false, false);
}

__device__ __forceinline__ float sigmoidf_(float x) { return 1.0f / (1.0f + __expf(-x)); }

// ---------------- Precompute kernels ----------------

__global__ void cvt_bf16_kernel(const float* __restrict__ src, bf16_t* __restrict__ dst, int n) {
  int i = blockIdx.x * blockDim.x + threadIdx.x;
  if (i < n) dst[i] = (bf16_t)src[i];
}

// x [B,T,H] f32 -> x_bf [(t*B+b), H] bf16
__global__ void x_transpose_bf16_kernel(const float* __restrict__ x, bf16_t* __restrict__ xb) {
  long i = (long)blockIdx.x * blockDim.x + threadIdx.x;
  const long total = (long)T * B * H;
  if (i >= total) return;
  int  k = (int)(i % H);
  long r = i / H;
  int  b = (int)(r % B);
  int  t = (int)(r / B);
  xb[i] = (bf16_t)x[((long)b * T + t) * H + k];
}

__global__ void init_state_kernel(const float* __restrict__ h0,
                                  float* __restrict__ h0s, float* __restrict__ h1s,
                                  bf16_t* __restrict__ h0b, bf16_t* __restrict__ h1b) {
  int i = blockIdx.x * blockDim.x + threadIdx.x;
  if (i >= B * H) return;
  float v = h0[i % H];
  h0s[i] = v; h1s[i] = v;
  h0b[i] = (bf16_t)v; h1b[i] = (bf16_t)v;
}

// Batched GEMM: C[M,N] = A[M,K](bf16) @ W[N,K]^T(bf16) + bias(+bias2).
// One wave computes a 64x64 output block (4x4 WMMA tiles, 16 accumulators = 128 VGPRs).
// __launch_bounds__(128,1): 4-wave workgroup at min occupancy so the register
// allocator keeps all accumulators resident (round-2 build spilled to scratch
// with the default budget). Row blocks of 64 == one timestep (B=64), so tshift
// maps the whole block's A rows to min(t+1,T-1)*B.
__global__ void __launch_bounds__(128, 1)
wmma_gemm64_kernel(const bf16_t* __restrict__ A, const bf16_t* __restrict__ W,
                   const float* __restrict__ bias, const float* __restrict__ bias2,
                   float* __restrict__ C, int Mblocks, int Nblocks, int Kdim, int tshift) {
  int wid  = (blockIdx.x * blockDim.x + threadIdx.x) >> 5;
  int lane = threadIdx.x & 31;
  int total = Mblocks * Nblocks;
  if (wid >= total) return;            // wave-uniform
  int mb = wid % Mblocks;
  int nb = wid / Mblocks;
  int r0 = mb * 64;
  int arow = r0;
  if (tshift) {                        // r0/B == mb exactly (64 rows per t)
    int t2 = (mb + 1 < T) ? (mb + 1) : (T - 1);
    arow = t2 * B;
  }
  const bf16_t* Abase = A + (long)arow * Kdim;
  const bf16_t* Wbase = W + (long)(nb * 64) * Kdim;

  v8f acc[4][4];
#pragma unroll
  for (int i = 0; i < 4; ++i)
#pragma unroll
    for (int j = 0; j < 4; ++j) acc[i][j] = (v8f){};

  for (int kk = 0; kk < Kdim; kk += 32) {
    __builtin_prefetch(Wbase + kk + 128, 0, 1);
    v16bf a[4];
#pragma unroll
    for (int i = 0; i < 4; ++i)
      a[i] = load_a_tile(Abase + (long)i * 16 * Kdim + kk, Kdim, lane);
#pragma unroll
    for (int j = 0; j < 4; ++j) {
      v16bf b = load_b_tile(Wbase + (long)j * 16 * Kdim + kk, Kdim, lane);
#pragma unroll
      for (int i = 0; i < 4; ++i)
        acc[i][j] = wmma_bf16(a[i], b, acc[i][j]);
    }
  }

  int hi  = lane >> 4;
  long Ncols = (long)Nblocks * 64;
#pragma unroll
  for (int j = 0; j < 4; ++j) {
    int col = nb * 64 + j * 16 + (lane & 15);
    float bs = (bias ? bias[col] : 0.f) + (bias2 ? bias2[col] : 0.f);
#pragma unroll
    for (int i = 0; i < 4; ++i) {
#pragma unroll
      for (int v = 0; v < 8; ++v) {
        long row = r0 + i * 16 + v + 8 * hi;
        C[row * Ncols + col] = acc[i][j][v] + bs;
      }
    }
  }
}

// ---------------- Per-timestep fused kernels ----------------

// Layer 0: gh = h0@W_hh0^T, fuse GRU nonlinearity with precomputed gi0[t]. 256 waves (4 x 64 tiles).
__global__ void gru_l0_step_kernel(const bf16_t* __restrict__ h0_bf,
                                   const bf16_t* __restrict__ Whh0,
                                   const float* __restrict__ gi0_t,
                                   const float* __restrict__ bhh0,
                                   const float* __restrict__ h0_state,
                                   float* __restrict__ h0_cand,
                                   bf16_t* __restrict__ h0_cand_bf) {
  int wid  = (blockIdx.x * blockDim.x + threadIdx.x) >> 5;
  int lane = threadIdx.x & 31;
  int mt = wid & 3, nt = wid >> 2;
  const bf16_t* Abase = h0_bf + mt * 16 * H;
  const bf16_t* Br = Whh0 + (nt * 16) * H;
  const bf16_t* Bz = Br + 1024 * H;
  const bf16_t* Bn = Bz + 1024 * H;
  v8f cr = {}, cz = {}, cn = {};
#pragma unroll 2
  for (int kk = 0; kk < H; kk += 32) {
    v16bf a = load_a_tile(Abase + kk, H, lane);
    cr = wmma_bf16(a, load_b_tile(Br + kk, H, lane), cr);
    cz = wmma_bf16(a, load_b_tile(Bz + kk, H, lane), cz);
    cn = wmma_bf16(a, load_b_tile(Bn + kk, H, lane), cn);
  }
  int col = nt * 16 + (lane & 15);
  int hi  = lane >> 4;
  float br = bhh0[col], bz = bhh0[1024 + col], bn = bhh0[2048 + col];
#pragma unroll
  for (int v = 0; v < 8; ++v) {
    int row = mt * 16 + v + 8 * hi;
    float gir = gi0_t[row * N3 + col];
    float giz = gi0_t[row * N3 + 1024 + col];
    float gin = gi0_t[row * N3 + 2048 + col];
    float r  = sigmoidf_(gir + cr[v] + br);
    float z  = sigmoidf_(giz + cz[v] + bz);
    float nn = tanhf(gin + r * (cn[v] + bn));
    float hp = h0_state[row * H + col];
    float hn = (1.f - z) * nn + z * hp;
    h0_cand[row * H + col]    = hn;
    h0_cand_bf[row * H + col] = (bf16_t)hn;
  }
}

// Layer 1: gi = h0_cand@W_ih1^T, gh = h1@W_hh1^T, fused nonlinearity. 6 accumulators per wave.
__global__ void gru_l1_step_kernel(const bf16_t* __restrict__ xin_bf,
                                   const bf16_t* __restrict__ h1_bf,
                                   const bf16_t* __restrict__ Wih1,
                                   const bf16_t* __restrict__ Whh1,
                                   const float* __restrict__ bih1,
                                   const float* __restrict__ bhh1,
                                   const float* __restrict__ h1_state,
                                   float* __restrict__ h1_cand,
                                   bf16_t* __restrict__ h1_cand_bf) {
  int wid  = (blockIdx.x * blockDim.x + threadIdx.x) >> 5;
  int lane = threadIdx.x & 31;
  int mt = wid & 3, nt = wid >> 2;
  const bf16_t* Ax = xin_bf + mt * 16 * H;
  const bf16_t* Ah = h1_bf  + mt * 16 * H;
  const bf16_t* BiR = Wih1 + (nt * 16) * H;
  const bf16_t* BiZ = BiR + 1024 * H;
  const bf16_t* BiN = BiZ + 1024 * H;
  const bf16_t* BhR = Whh1 + (nt * 16) * H;
  const bf16_t* BhZ = BhR + 1024 * H;
  const bf16_t* BhN = BhZ + 1024 * H;
  v8f ir = {}, iz = {}, in_ = {}, hr = {}, hz = {}, hn = {};
  for (int kk = 0; kk < H; kk += 32) {
    v16bf a1 = load_a_tile(Ax + kk, H, lane);
    v16bf a2 = load_a_tile(Ah + kk, H, lane);
    ir  = wmma_bf16(a1, load_b_tile(BiR + kk, H, lane), ir);
    iz  = wmma_bf16(a1, load_b_tile(BiZ + kk, H, lane), iz);
    in_ = wmma_bf16(a1, load_b_tile(BiN + kk, H, lane), in_);
    hr  = wmma_bf16(a2, load_b_tile(BhR + kk, H, lane), hr);
    hz  = wmma_bf16(a2, load_b_tile(BhZ + kk, H, lane), hz);
    hn  = wmma_bf16(a2, load_b_tile(BhN + kk, H, lane), hn);
  }
  int col = nt * 16 + (lane & 15);
  int hi  = lane >> 4;
  float bir = bih1[col], biz = bih1[1024 + col], bin = bih1[2048 + col];
  float bhr = bhh1[col], bhz = bhh1[1024 + col], bhn = bhh1[2048 + col];
#pragma unroll
  for (int v = 0; v < 8; ++v) {
    int row = mt * 16 + v + 8 * hi;
    float r  = sigmoidf_(ir[v] + bir + hr[v] + bhr);
    float z  = sigmoidf_(iz[v] + biz + hz[v] + bhz);
    float nn = tanhf(in_[v] + bin + r * (hn[v] + bhn));
    float hp = h1_state[row * H + col];
    float hv = (1.f - z) * nn + z * hp;
    h1_cand[row * H + col]    = hv;
    h1_cand_bf[row * H + col] = (bf16_t)hv;
  }
}

// Gate GEMM: y = gateX[t] + h1_cand @ gW2^T. 256 waves.
__global__ void gate_gemm_kernel(const bf16_t* __restrict__ h1c_bf,
                                 const bf16_t* __restrict__ gW2b,
                                 const float* __restrict__ gateX_t,
                                 float* __restrict__ y) {
  int wid  = (blockIdx.x * blockDim.x + threadIdx.x) >> 5;
  int lane = threadIdx.x & 31;
  int mt = wid & 3, nt = wid >> 2;
  const bf16_t* Abase = h1c_bf + mt * 16 * H;
  const bf16_t* Bb    = gW2b + (nt * 16) * H;
  v8f c = {};
#pragma unroll 2
  for (int kk = 0; kk < H; kk += 32)
    c = wmma_bf16(load_a_tile(Abase + kk, H, lane), load_b_tile(Bb + kk, H, lane), c);
  int col = nt * 16 + (lane & 15);
  int hi  = lane >> 4;
#pragma unroll
  for (int v = 0; v < 8; ++v) {
    int row = mt * 16 + v + 8 * hi;
    y[row * H + col] = gateX_t[row * H + col] + c[v];
  }
}

// Gate finalize: per-batch dot(y, gWF), sigmoid, masking, reset, state commit + bf16 re-encode.
__global__ void gate_finalize_kernel(const float* __restrict__ y, const float* __restrict__ gWF,
                                     const float* __restrict__ gbF, const int* __restrict__ lengths,
                                     int t, const float* __restrict__ h0vec,
                                     const float* __restrict__ h0_cand, const float* __restrict__ h1_cand,
                                     float* __restrict__ h0_state, float* __restrict__ h1_state,
                                     bf16_t* __restrict__ h0_bf, bf16_t* __restrict__ h1_bf,
                                     float* __restrict__ out, float* __restrict__ gate_out) {
  __shared__ float red[256];
  __shared__ float gsh;
  int b = blockIdx.x, tid = threadIdx.x;
  float acc = 0.f;
  for (int j = tid; j < H; j += 256) acc += y[b * H + j] * gWF[j];
  red[tid] = acc;
  __syncthreads();
  for (int s = 128; s > 0; s >>= 1) {
    if (tid < s) red[tid] += red[tid + s];
    __syncthreads();
  }
  if (tid == 0) gsh = sigmoidf_(red[0] + gbF[0]);
  __syncthreads();
  int  len    = lengths[b];
  bool valid  = t < len;
  bool islast = (t == len - 1);
  float gt    = islast ? 1.0f : gsh;
  bool reset  = valid && (gt > 0.5f);
  for (int j = tid; j < H; j += 256) {
    float oc = h1_cand[b * H + j];
    out[((long)b * T + t) * H + j] = valid ? oc : 0.f;
    float n0 = valid ? (reset ? h0vec[j] : h0_cand[b * H + j]) : h0_state[b * H + j];
    float n1 = valid ? (reset ? h0vec[j] : oc)                 : h1_state[b * H + j];
    h0_state[b * H + j] = n0;
    h1_state[b * H + j] = n1;
    h0_bf[b * H + j] = (bf16_t)n0;
    h1_bf[b * H + j] = (bf16_t)n1;
  }
  if (tid == 0) gate_out[b * T + t] = valid ? gt : 0.f;
}

// ---------------- Host launch ----------------

extern "C" void kernel_launch(void* const* d_in, const int* in_sizes, int n_in,
                              void* d_out, int out_size, void* d_ws, size_t ws_size,
                              hipStream_t stream) {
  const float* x       = (const float*)d_in[0];
  const float* h0      = (const float*)d_in[1];
  const int*   lengths = (const int*)d_in[2];
  const float* W_ih    = (const float*)d_in[3];
  const float* W_hh    = (const float*)d_in[4];
  const float* b_ih    = (const float*)d_in[5];
  const float* b_hh    = (const float*)d_in[6];
  const float* gW1     = (const float*)d_in[7];
  const float* gb1     = (const float*)d_in[8];
  const float* gW2     = (const float*)d_in[9];
  const float* gb2     = (const float*)d_in[10];
  const float* gWF     = (const float*)d_in[11];
  const float* gbF     = (const float*)d_in[12];

  float* out      = (float*)d_out;
  float* gate_out = out + (size_t)B * T * H;

  char* p = (char*)d_ws;
  auto alloc = [&](size_t bytes) { char* q = p; p += (bytes + 255) & ~(size_t)255; return q; };

  bf16_t* x_bf    = (bf16_t*)alloc((size_t)T * B * H * 2);
  bf16_t* Wih0_bf = (bf16_t*)alloc((size_t)N3 * H * 2);
  bf16_t* Wih1_bf = (bf16_t*)alloc((size_t)N3 * H * 2);
  bf16_t* Whh0_bf = (bf16_t*)alloc((size_t)N3 * H * 2);
  bf16_t* Whh1_bf = (bf16_t*)alloc((size_t)N3 * H * 2);
  bf16_t* gW1_bf  = (bf16_t*)alloc((size_t)H * H * 2);
  bf16_t* gW2_bf  = (bf16_t*)alloc((size_t)H * H * 2);
  float*  gi0_all = (float*)alloc((size_t)T * B * N3 * 4);
  float*  gateX   = (float*)alloc((size_t)T * B * H * 4);
  float*  h0s  = (float*)alloc((size_t)B * H * 4);
  float*  h1s  = (float*)alloc((size_t)B * H * 4);
  float*  h0c  = (float*)alloc((size_t)B * H * 4);
  float*  h1c  = (float*)alloc((size_t)B * H * 4);
  bf16_t* h0bf  = (bf16_t*)alloc((size_t)B * H * 2);
  bf16_t* h1bf  = (bf16_t*)alloc((size_t)B * H * 2);
  bf16_t* h0cbf = (bf16_t*)alloc((size_t)B * H * 2);
  bf16_t* h1cbf = (bf16_t*)alloc((size_t)B * H * 2);
  float*  ybuf  = (float*)alloc((size_t)B * H * 4);

  const int nW3 = N3 * H, nW1 = H * H;
  cvt_bf16_kernel<<<(nW3 + 255) / 256, 256, 0, stream>>>(W_ih,       Wih0_bf, nW3);
  cvt_bf16_kernel<<<(nW3 + 255) / 256, 256, 0, stream>>>(W_ih + nW3, Wih1_bf, nW3);
  cvt_bf16_kernel<<<(nW3 + 255) / 256, 256, 0, stream>>>(W_hh,       Whh0_bf, nW3);
  cvt_bf16_kernel<<<(nW3 + 255) / 256, 256, 0, stream>>>(W_hh + nW3, Whh1_bf, nW3);
  cvt_bf16_kernel<<<(nW1 + 255) / 256, 256, 0, stream>>>(gW1, gW1_bf, nW1);
  cvt_bf16_kernel<<<(nW1 + 255) / 256, 256, 0, stream>>>(gW2, gW2_bf, nW1);

  {
    long totalX = (long)T * B * H;
    x_transpose_bf16_kernel<<<(int)((totalX + 255) / 256), 256, 0, stream>>>(x, x_bf);
  }
  init_state_kernel<<<(B * H + 255) / 256, 256, 0, stream>>>(h0, h0s, h1s, h0bf, h1bf);

  {   // gi0_all[t*B+b, 3H] = x @ W_ih0^T + b_ih0   (batched over all timesteps)
    int Mblocks = T * B / 64, Nblocks = N3 / 64;
    int waves   = Mblocks * Nblocks;
    wmma_gemm64_kernel<<<(waves + 3) / 4, 128, 0, stream>>>(x_bf, Wih0_bf, b_ih, nullptr,
                                                            gi0_all, Mblocks, Nblocks, H, 0);
  }
  {   // gateX[t*B+b, H] = x_{t+1} @ gW1^T + gb1 + gb2
    int Mblocks = T * B / 64, Nblocks = H / 64;
    int waves   = Mblocks * Nblocks;
    wmma_gemm64_kernel<<<(waves + 3) / 4, 128, 0, stream>>>(x_bf, gW1_bf, gb1, gb2,
                                                            gateX, Mblocks, Nblocks, H, 1);
  }

  for (int t = 0; t < T; ++t) {
    gru_l0_step_kernel<<<32, 256, 0, stream>>>(h0bf, Whh0_bf, gi0_all + (size_t)t * B * N3,
                                               b_hh, h0s, h0c, h0cbf);
    gru_l1_step_kernel<<<32, 256, 0, stream>>>(h0cbf, h1bf, Wih1_bf, Whh1_bf,
                                               b_ih + N3, b_hh + N3, h1s, h1c, h1cbf);
    gate_gemm_kernel<<<32, 256, 0, stream>>>(h1cbf, gW2_bf, gateX + (size_t)t * B * H, ybuf);
    gate_finalize_kernel<<<B, 256, 0, stream>>>(ybuf, gWF, gbF, lengths, t, h0,
                                                h0c, h1c, h0s, h1s, h0bf, h1bf,
                                                out, gate_out);
  }
}
